// FeedBack_22548578304023
// MI455X (gfx1250) — matrix operand order repaired and checked by reading, not compiled, not run on hardware
//
#include <hip/hip_runtime.h>
#include <hip/hip_bf16.h>

// ---------------------------------------------------------------------------
// RNN forecaster on gfx1250: bf16 WMMA (v_wmma_f32_16x16x32_bf16) GEMMs with
// fused bias+tanh, h state ping-ponged through L2-resident workspace.
// Wave tile 32x64 (2x4 accumulators) -> 8 WMMAs per 12 b128 loads per k-chunk.
// ---------------------------------------------------------------------------

typedef __bf16 bf16_t;
typedef __attribute__((ext_vector_type(8)))  __bf16 v8bf;
typedef __attribute__((ext_vector_type(16))) __bf16 v16bf;
typedef __attribute__((ext_vector_type(8)))  float  v8f;

#define NBATCH   16384
#define NFEAT    19
#define NUNITS   512
#define NSTEPS   24
#define KPAD     32          // padded feature block (19 -> 32)
#define LDA      544         // KPAD + NUNITS, activation row stride (elems)
#define NKCHUNK  17          // 544 / 32 k-chunks for the recurrent GEMM

union Frag16 { v16bf v; v8bf h[2]; };

// ---------------------------------------------------------------------------
// Weight prep: Wt[n=512][k=544] = [K^T | 0 | R^T] bf16 ; Wdt[f=32][u=512] = W^T
// ---------------------------------------------------------------------------
__global__ void prep_weights(const float* __restrict__ Kmat,   // [19][512]
                             const float* __restrict__ Rmat,   // [512][512]
                             const float* __restrict__ Wmat,   // [512][19]
                             bf16_t* __restrict__ Wt,
                             bf16_t* __restrict__ Wdt)
{
    int idx = blockIdx.x * blockDim.x + threadIdx.x;
    const int totalWt = NUNITS * LDA;            // 278528
    if (idx < totalWt) {
        int n = idx / LDA, k = idx % LDA;
        float v = 0.0f;
        if (k < NFEAT)       v = Kmat[k * NUNITS + n];
        else if (k >= KPAD)  v = Rmat[(k - KPAD) * NUNITS + n];
        Wt[idx] = (bf16_t)v;
    } else {
        int j = idx - totalWt;
        if (j < KPAD * NUNITS) {
            int f = j / NUNITS, u = j % NUNITS;
            float v = (f < NFEAT) ? Wmat[u * NFEAT + f] : 0.0f;
            Wdt[j] = (bf16_t)v;
        }
    }
}

// Zero a buffer in 16-byte chunks (h0 = 0 initialization)
__global__ void zero_buf(uint4* __restrict__ p, int n16)
{
    int i = blockIdx.x * blockDim.x + threadIdx.x;
    if (i < n16) p[i] = make_uint4(0u, 0u, 0u, 0u);
}

// Copy x_t (f32 [B,24,19]) into the padded x-slot of an activation buffer
__global__ void set_x(const float* __restrict__ inputs, int t,
                      bf16_t* __restrict__ Abuf)
{
    int idx = blockIdx.x * blockDim.x + threadIdx.x;   // over NBATCH*32
    int b = idx >> 5, k = idx & 31;
    if (b < NBATCH) {
        float v = (k < NFEAT) ? inputs[(b * NSTEPS + t) * NFEAT + k] : 0.0f;
        Abuf[(size_t)b * LDA + k] = (bf16_t)v;
    }
}

// ---------------------------------------------------------------------------
// Recurrent GEMM: Hout_h = tanh(A[B x 544] @ Wt^T + bias), bf16 in/out, f32 acc
// Block = 256 thr (8 waves), block tile 128x128; wave tile 32x64 (2x4 accums).
// ---------------------------------------------------------------------------
__global__ __launch_bounds__(256)
void rnn_gemm(const bf16_t* __restrict__ A,    // [NBATCH][LDA]
              const bf16_t* __restrict__ Wt,   // [NUNITS][LDA] (n-major)
              const float*  __restrict__ bias, // [NUNITS]
              bf16_t* __restrict__ Hout)       // dst Abuf base; h at col KPAD
{
    const int lane = threadIdx.x & 31;
    const int wave = threadIdx.x >> 5;
    const int wm   = wave & 3;                  // 0..3
    const int wn   = wave >> 2;                 // 0..1
    const int half = lane >> 4;
    const int l16  = lane & 15;

    const int rm = blockIdx.x * 128 + wm * 32;  // row base of this wave
    const int cn = blockIdx.y * 128 + wn * 64;  // col base of this wave

    v8f acc[2][4];
    #pragma unroll
    for (int nt = 0; nt < 4; ++nt) {
        float bv = bias[cn + nt * 16 + l16];
        #pragma unroll
        for (int mt = 0; mt < 2; ++mt)
            #pragma unroll
            for (int r = 0; r < 8; ++r) acc[mt][nt][r] = bv;
    }

    for (int kc = 0; kc < NKCHUNK; ++kc) {
        const int kof = kc * 32;
        Frag16 fa[2], fb[4];
        // A fragments: a[e] = A[m][kof + 16*(e/8) + 8*half + e%8]
        #pragma unroll
        for (int mt = 0; mt < 2; ++mt) {
            const bf16_t* p = A + (size_t)(rm + mt * 16 + l16) * LDA + kof + half * 8;
            fa[mt].h[0] = *(const v8bf*)(p);
            fa[mt].h[1] = *(const v8bf*)(p + 16);
        }
        // B fragments: b[e] = Wt[n][kof + 16*(e/8) + 8*half + e%8]
        #pragma unroll
        for (int nt = 0; nt < 4; ++nt) {
            const bf16_t* q = Wt + (size_t)(cn + nt * 16 + l16) * LDA + kof + half * 8;
            fb[nt].h[0] = *(const v8bf*)(q);
            fb[nt].h[1] = *(const v8bf*)(q + 16);
        }
        #pragma unroll
        for (int mt = 0; mt < 2; ++mt)
            #pragma unroll
            for (int nt = 0; nt < 4; ++nt)
                acc[mt][nt] = __builtin_amdgcn_wmma_f32_16x16x32_bf16(
                    false, fa[mt].v, false, fb[nt].v,
                    (short)0, acc[mt][nt], false, false);
    }

    // Epilogue: tanh, convert to bf16, scatter into dst h-slot
    #pragma unroll
    for (int mt = 0; mt < 2; ++mt)
        #pragma unroll
        for (int nt = 0; nt < 4; ++nt)
            #pragma unroll
            for (int r = 0; r < 8; ++r) {
                float v = tanhf(acc[mt][nt][r]);
                int row = rm + mt * 16 + half * 8 + r;
                int col = cn + nt * 16 + l16;
                Hout[(size_t)row * LDA + KPAD + col] = (bf16_t)v;
            }
}

// ---------------------------------------------------------------------------
// Prediction GEMM: pred = h @ W + b2 ; writes f32 to d_out[:,step,:] and bf16
// pred into the x-slot of the same activation buffer (feedback).
// Block = 256 thr (8 waves), each wave: 16 rows x 32 cols (2 WMMA accums).
// ---------------------------------------------------------------------------
__global__ __launch_bounds__(256)
void rnn_pred(const bf16_t* __restrict__ Abuf, // h at col KPAD, stride LDA
              const bf16_t* __restrict__ Wdt,  // [32][512] (f-major)
              const float*  __restrict__ db,   // [19]
              float* __restrict__ out,         // d_out + step*19, row stride 24*19
              bf16_t* __restrict__ predslot)   // Abuf x-slot (cols 0..31)
{
    const int lane = threadIdx.x & 31;
    const int wave = threadIdx.x >> 5;
    const int half = lane >> 4;
    const int l16  = lane & 15;

    const int rb = blockIdx.x * 128 + wave * 16;

    v8f acc[2];
    #pragma unroll
    for (int nt = 0; nt < 2; ++nt) {
        int f = nt * 16 + l16;
        float bv = (f < NFEAT) ? db[f] : 0.0f;
        #pragma unroll
        for (int r = 0; r < 8; ++r) acc[nt][r] = bv;
    }

    for (int kc = 0; kc < NUNITS / 32; ++kc) {
        const int kof = kc * 32;
        Frag16 fa, fb[2];
        const bf16_t* p = Abuf + (size_t)(rb + l16) * LDA + KPAD + kof + half * 8;
        fa.h[0] = *(const v8bf*)(p);
        fa.h[1] = *(const v8bf*)(p + 16);
        #pragma unroll
        for (int nt = 0; nt < 2; ++nt) {
            const bf16_t* q = Wdt + (size_t)(nt * 16 + l16) * NUNITS + kof + half * 8;
            fb[nt].h[0] = *(const v8bf*)(q);
            fb[nt].h[1] = *(const v8bf*)(q + 16);
        }
        #pragma unroll
        for (int nt = 0; nt < 2; ++nt)
            acc[nt] = __builtin_amdgcn_wmma_f32_16x16x32_bf16(
                false, fa.v, false, fb[nt].v,
                (short)0, acc[nt], false, false);
    }

    #pragma unroll
    for (int nt = 0; nt < 2; ++nt)
        #pragma unroll
        for (int r = 0; r < 8; ++r) {
            int f   = nt * 16 + l16;
            int row = rb + half * 8 + r;
            float v = acc[nt][r];
            if (f < NFEAT)
                out[(size_t)row * (NSTEPS * NFEAT) + f] = v;
            predslot[(size_t)row * LDA + f] = (bf16_t)v;  // f>=19 -> exact 0
        }
}

// ---------------------------------------------------------------------------
extern "C" void kernel_launch(void* const* d_in, const int* in_sizes, int n_in,
                              void* d_out, int out_size, void* d_ws, size_t ws_size,
                              hipStream_t stream)
{
    const float* inputs = (const float*)d_in[0];   // [16384,24,19]
    const float* kmat   = (const float*)d_in[1];   // [19,512]
    const float* rmat   = (const float*)d_in[2];   // [512,512]
    const float* bias   = (const float*)d_in[3];   // [512]
    const float* dw     = (const float*)d_in[4];   // [512,19]
    const float* db     = (const float*)d_in[5];   // [19]
    float* out = (float*)d_out;                    // [16384,24,19]

    // Workspace carve-up (all offsets 256B aligned)
    char* ws = (char*)d_ws;
    const size_t wt_bytes  = (size_t)NUNITS * LDA * sizeof(bf16_t);   // 557056
    const size_t wdt_bytes = (size_t)KPAD * NUNITS * sizeof(bf16_t);  // 32768
    const size_t ab_bytes  = (size_t)NBATCH * LDA * sizeof(bf16_t);   // 17825792
    bf16_t* Wt  = (bf16_t*)(ws);
    bf16_t* Wdt = (bf16_t*)(ws + wt_bytes);
    bf16_t* Ab0 = (bf16_t*)(ws + wt_bytes + wdt_bytes);
    bf16_t* Ab1 = (bf16_t*)(ws + wt_bytes + wdt_bytes + ab_bytes);
    bf16_t* Ab[2] = {Ab0, Ab1};

    // 1) weights -> bf16, transposed+padded
    {
        int total = NUNITS * LDA + KPAD * NUNITS;
        prep_weights<<<(total + 255) / 256, 256, 0, stream>>>(kmat, rmat, dw, Wt, Wdt);
    }
    // 2) h0 = 0 (zero whole Ab0; x-slot overwritten by set_x(t=0))
    {
        int n16 = (int)(ab_bytes / 16);
        zero_buf<<<(n16 + 255) / 256, 256, 0, stream>>>((uint4*)Ab0, n16);
    }

    dim3 ggrid(NBATCH / 128, NUNITS / 128);  // 128 x 4

    // 3) warmup: 24 RNN steps over the input window
    for (int t = 0; t < NSTEPS; ++t) {
        set_x<<<(NBATCH * 32) / 256, 256, 0, stream>>>(inputs, t, Ab[t & 1]);
        rnn_gemm<<<ggrid, 256, 0, stream>>>(Ab[t & 1], Wt, bias, Ab[(t + 1) & 1]);
    }

    // 4) first prediction (h lives in Ab[0] after 24 steps)
    rnn_pred<<<NBATCH / 128, 256, 0, stream>>>(Ab[0], Wdt, db, out + 0 * NFEAT, Ab[0]);

    // 5) autoregressive feedback: 23 more (gemm, pred) pairs
    for (int i = 1; i < NSTEPS; ++i) {
        bf16_t* src = Ab[(i + 1) & 1];
        bf16_t* dst = Ab[i & 1];
        rnn_gemm<<<ggrid, 256, 0, stream>>>(src, Wt, bias, dst);
        rnn_pred<<<NBATCH / 128, 256, 0, stream>>>(dst, Wdt, db, out + i * NFEAT, dst);
    }
}
